// CausalSelfAttention_10943576670636
// MI455X (gfx1250) — matrix operand order repaired and checked by reading, not compiled, not run on hardware
//
#include <hip/hip_runtime.h>
#include <math.h>

typedef __bf16 bf16_t;
typedef __attribute__((ext_vector_type(8)))  __bf16 bf16x8;
typedef __attribute__((ext_vector_type(16))) __bf16 bf16x16;
typedef __attribute__((ext_vector_type(8)))  float  f32x8;

#define B_  4
#define T_  2048
#define C_  1024
#define H_  16
#define D_  64
#define C3_ 3072

static __device__ __forceinline__ bf16x16 make_frag(bf16x8 lo, bf16x8 hi) {
  return __builtin_shufflevector(lo, hi, 0,1,2,3,4,5,6,7,8,9,10,11,12,13,14,15);
}

// A fragment: 16x32 bf16 tile from row-major source, leading dim ld (elems).
// ISA layout: lanes 0-15 (M=0..15): V0-3 = K 0..7, V4-7 = K 16..23;
//             lanes 16-31 (M=0..15): V0-3 = K 8..15, V4-7 = K 24..31.
static __device__ __forceinline__ bf16x16 load_frag_a(const bf16_t* base, int ld,
                                                      int row0, int k0, int lane) {
  int m  = lane & 15;
  int kh = lane >> 4;
  const bf16_t* p = base + (size_t)(row0 + m) * ld + k0;
  bf16x8 lo = *(const bf16x8*)(p + kh * 8);
  bf16x8 hi = *(const bf16x8*)(p + 16 + kh * 8);
  return make_frag(lo, hi);
}

// B fragment: 32x16 (KxN) bf16, source stored N-major [n][k] with leading dim ld.
// ISA layout: lanes 0-15: K 0..7, 16..23; lanes 16-31: K 8..15, 24..31; col = lane&15.
static __device__ __forceinline__ bf16x16 load_frag_b(const bf16_t* baseT, int ld,
                                                      int n0, int k0, int lane) {
  int n  = lane & 15;
  int kh = lane >> 4;
  const bf16_t* p = baseT + (size_t)(n0 + n) * ld + k0;
  bf16x8 lo = *(const bf16x8*)(p + kh * 8);
  bf16x8 hi = *(const bf16x8*)(p + 16 + kh * 8);
  return make_frag(lo, hi);
}

static __device__ __forceinline__ f32x8 wmma_bf16(bf16x16 a, bf16x16 b, f32x8 c) {
  return __builtin_amdgcn_wmma_f32_16x16x32_bf16(false, a, false, b, (short)0, c, false, false);
}

// gfx1250 async DMA: global -> LDS, 16B per lane, tracked by ASYNCcnt.
static __device__ __forceinline__ void async_copy_b128(unsigned lds_off, const bf16_t* gp) {
  asm volatile("global_load_async_to_lds_b128 %0, %1, off"
               :: "v"(lds_off), "v"((unsigned long long)(uintptr_t)gp)
               : "memory");
}

// ---------------- precision / layout prep ----------------

__global__ void __launch_bounds__(256) cast_f32_to_bf16(const float* __restrict__ in,
                                                        bf16_t* __restrict__ out, int n) {
  int i = blockIdx.x * blockDim.x + threadIdx.x;
  int stride = gridDim.x * blockDim.x;
  for (; i < n; i += stride) out[i] = (bf16_t)in[i];
}

// in[rows][cols] (f32) -> out[cols][rows] (bf16)
__global__ void __launch_bounds__(256) transpose_cast(const float* __restrict__ in,
                                                      bf16_t* __restrict__ out,
                                                      int rows, int cols) {
  int i = blockIdx.x * blockDim.x + threadIdx.x;
  int stride = gridDim.x * blockDim.x;
  int n = rows * cols;
  for (; i < n; i += stride) {
    int k = i / cols, c = i - k * cols;
    out[(size_t)c * rows + k] = (bf16_t)in[i];
  }
}

// ---------------- QKV GEMM: 128x64 block tile, B staged in LDS via async DMA ----------

// Block = 256 threads = 8 waves; wave w owns rows [w*16, w*16+16) of the block's 128 rows.
// B tile (64 cols x 64 k, n-major) double-buffered in LDS, filled by
// global_load_async_to_lds_b128 (2 instructions per wave per chunk).
__global__ void __launch_bounds__(256) gemm_qkv(const bf16_t* __restrict__ xb,
                                                const bf16_t* __restrict__ wT,
                                                const float* __restrict__ bias,
                                                bf16_t* __restrict__ q,
                                                bf16_t* __restrict__ k,
                                                bf16_t* __restrict__ vT) {
  __shared__ __align__(16) bf16_t Bs[2][64 * 64];   // 2 x 8KB
  const int NTB = C3_ / 64;                         // 48 N-blocks
  int mtb = blockIdx.x / NTB, ntb = blockIdx.x - mtb * NTB;
  int n0 = ntb * 64;
  int tid  = threadIdx.x;
  int lane = tid & 31;
  int wave = tid >> 5;
  int row0 = mtb * 128 + wave * 16;

  // this thread's two 16B segments of the 8KB B tile (512 segments total)
  int seg0 = tid, seg1 = tid + 256;
  const bf16_t* g0 = wT + (size_t)(n0 + (seg0 >> 3)) * C_ + (seg0 & 7) * 8;
  const bf16_t* g1 = wT + (size_t)(n0 + (seg1 >> 3)) * C_ + (seg1 & 7) * 8;
  unsigned l0 = (unsigned)((seg0 >> 3) * 128 + (seg0 & 7) * 16);
  unsigned l1 = (unsigned)((seg1 >> 3) * 128 + (seg1 & 7) * 16);
  unsigned bsbase = (unsigned)(uintptr_t)(&Bs[0][0]);   // LDS byte offset
  const unsigned bufoff[2] = {bsbase, bsbase + 64u * 64u * 2u};

  const int NCH = C_ / 64;   // 16 K-chunks
  // prologue: chunk 0 -> buffer 0
  async_copy_b128(bufoff[0] + l0, g0);
  async_copy_b128(bufoff[0] + l1, g1);

  f32x8 acc0 = {}, acc1 = {}, acc2 = {}, acc3 = {};
  for (int ch = 0; ch < NCH; ++ch) {
    int cur = ch & 1;
    if (ch + 1 < NCH) {
      const bf16_t* ng0 = g0 + (ch + 1) * 64;
      const bf16_t* ng1 = g1 + (ch + 1) * 64;
      async_copy_b128(bufoff[cur ^ 1] + l0, ng0);
      async_copy_b128(bufoff[cur ^ 1] + l1, ng1);
      asm volatile("s_wait_asynccnt 0x2" ::: "memory");   // chunk ch landed
    } else {
      asm volatile("s_wait_asynccnt 0x0" ::: "memory");
    }
    __syncthreads();                                      // all waves' DMA done

    const bf16_t* bs = &Bs[cur][0];
    #pragma unroll
    for (int ko = 0; ko < 64; ko += 32) {
      bf16x16 a = load_frag_a(xb, C_, row0, ch * 64 + ko, lane);
      acc0 = wmma_bf16(a, load_frag_b(bs, 64,  0, ko, lane), acc0);
      acc1 = wmma_bf16(a, load_frag_b(bs, 64, 16, ko, lane), acc1);
      acc2 = wmma_bf16(a, load_frag_b(bs, 64, 32, ko, lane), acc2);
      acc3 = wmma_bf16(a, load_frag_b(bs, 64, 48, ko, lane), acc3);
    }
    __syncthreads();                                      // safe to overwrite buf[cur]
  }

  int nlo = lane & 15, lh = lane >> 4;
  f32x8 accs[4] = {acc0, acc1, acc2, acc3};
  #pragma unroll
  for (int j = 0; j < 4; ++j) {
    int col = n0 + j * 16 + nlo;
    int region = col / C_;
    int cc = col - region * C_;
    int h = cc >> 6, d = cc & 63;
    float bv = bias[col];
    #pragma unroll
    for (int r = 0; r < 8; ++r) {
      int row = row0 + r + lh * 8;           // D layout: lanes16-31 => M=r+8
      int bb = row >> 11, t = row & (T_ - 1);
      bf16_t o = (bf16_t)(accs[j][r] + bv);
      size_t bh = (size_t)(bb * H_ + h);
      if (region == 0)      q [(bh * T_ + t) * D_ + d] = o;
      else if (region == 1) k [(bh * T_ + t) * D_ + d] = o;
      else                  vT[(bh * D_ + d) * T_ + t] = o;   // V stored [B,H,D,T]
    }
  }
}

// ---------------- flash attention: one wave per 16-row Q tile ----------------

__global__ void __launch_bounds__(128) attn_flash(const bf16_t* __restrict__ q,
                                                  const bf16_t* __restrict__ kk,
                                                  const bf16_t* __restrict__ vT,
                                                  bf16_t* __restrict__ y) {
  __shared__ __align__(16) bf16_t Plds[4][16 * 32];
  int lane = threadIdx.x & 31;
  int wave = threadIdx.x >> 5;
  int gw = blockIdx.x * 4 + wave;
  int bh = gw >> 7;                 // (b*h) index, 0..63
  int qt = gw & 127;                // q tile within head
  int b = bh >> 4, h = bh & 15;
  const bf16_t* qh = q  + (size_t)bh * T_ * D_;
  const bf16_t* kh = kk + (size_t)bh * T_ * D_;
  const bf16_t* vh = vT + (size_t)bh * D_ * T_;
  bf16_t* Pw = Plds[wave];

  int qrow0 = qt * 16;
  int nlo = lane & 15, lh = lane >> 4;

  bf16x16 aq0 = load_frag_a(qh, D_, qrow0,  0, lane);
  bf16x16 aq1 = load_frag_a(qh, D_, qrow0, 32, lane);

  float mrow[8], lrow[8];
  #pragma unroll
  for (int r = 0; r < 8; ++r) { mrow[r] = -INFINITY; lrow[r] = 0.f; }
  f32x8 o0 = {}, o1 = {}, o2 = {}, o3 = {};

  const float scale = 0.125f;   // 1/sqrt(64)
  for (int kb = 0; kb <= qrow0 + 15; kb += 32) {
    if (kb + 32 <= qrow0 + 15) {  // pull next K tile toward the caches
      __builtin_prefetch(kh + (size_t)(kb + 32) * D_, 0, 0);
    }
    // S[16x32] = Q(16x64) @ K^T : contraction over d, K rows are contiguous [t][d]
    f32x8 s0 = {}, s1 = {};
    s0 = wmma_bf16(aq0, load_frag_b(kh, D_, kb,       0, lane), s0);
    s0 = wmma_bf16(aq1, load_frag_b(kh, D_, kb,      32, lane), s0);
    s1 = wmma_bf16(aq0, load_frag_b(kh, D_, kb + 16,  0, lane), s1);
    s1 = wmma_bf16(aq1, load_frag_b(kh, D_, kb + 16, 32, lane), s1);

    float sv0[8], sv1[8];
    bool edge = (kb + 31 > qrow0);
    #pragma unroll
    for (int r = 0; r < 8; ++r) {
      int row = qrow0 + r + lh * 8;
      float a0 = s0[r] * scale;
      float a1 = s1[r] * scale;
      if (edge) {
        if (kb + nlo      > row) a0 = -INFINITY;
        if (kb + 16 + nlo > row) a1 = -INFINITY;
      }
      sv0[r] = a0; sv1[r] = a1;
    }

    // online softmax; each row lives across 16 lanes of one half-wave
    #pragma unroll
    for (int r = 0; r < 8; ++r) {
      float rm = fmaxf(sv0[r], sv1[r]);
      rm = fmaxf(rm, __shfl_xor(rm, 8, 32));
      rm = fmaxf(rm, __shfl_xor(rm, 4, 32));
      rm = fmaxf(rm, __shfl_xor(rm, 2, 32));
      rm = fmaxf(rm, __shfl_xor(rm, 1, 32));
      float mnew = fmaxf(mrow[r], rm);
      float p0 = __expf(sv0[r] - mnew);
      float p1 = __expf(sv1[r] - mnew);
      float ps = p0 + p1;
      ps += __shfl_xor(ps, 8, 32);
      ps += __shfl_xor(ps, 4, 32);
      ps += __shfl_xor(ps, 2, 32);
      ps += __shfl_xor(ps, 1, 32);
      float alpha = __expf(mrow[r] - mnew);
      lrow[r] = lrow[r] * alpha + ps;
      mrow[r] = mnew;
      o0[r] *= alpha; o1[r] *= alpha; o2[r] *= alpha; o3[r] *= alpha;
      int prow = r + lh * 8;
      Pw[prow * 32 + nlo]      = (bf16_t)p0;   // P tile [16 rows][32 keys] in LDS
      Pw[prow * 32 + 16 + nlo] = (bf16_t)p1;
    }

    // same-wave cross-lane LDS dependency: DS ops are in-order; stop compiler reorder
    asm volatile("s_wait_dscnt 0x0" ::: "memory");

    // reload P as an A-fragment (16x32, K = 32 keys)
    const bf16_t* pp = Pw + (lane & 15) * 32 + lh * 8;
    bf16x8 plo = *(const bf16x8*)(pp);
    bf16x8 phi = *(const bf16x8*)(pp + 16);
    bf16x16 ap = make_frag(plo, phi);

    // O[16x64] += P @ V : V stored transposed [d][t] so B-frags are contiguous
    o0 = wmma_bf16(ap, load_frag_b(vh, T_,  0, kb, lane), o0);
    o1 = wmma_bf16(ap, load_frag_b(vh, T_, 16, kb, lane), o1);
    o2 = wmma_bf16(ap, load_frag_b(vh, T_, 32, kb, lane), o2);
    o3 = wmma_bf16(ap, load_frag_b(vh, T_, 48, kb, lane), o3);
    asm volatile("" ::: "memory");
  }

  f32x8 os[4] = {o0, o1, o2, o3};
  #pragma unroll
  for (int j = 0; j < 4; ++j) {
    #pragma unroll
    for (int r = 0; r < 8; ++r) {
      int row = qrow0 + r + lh * 8;
      float val = os[j][r] / lrow[r];
      y[((size_t)b * T_ + row) * C_ + h * D_ + j * 16 + nlo] = (bf16_t)val;
    }
  }
}

// ---------------- output projection: same LDS-staged GEMM, f32 out ----------------

__global__ void __launch_bounds__(256) gemm_proj(const bf16_t* __restrict__ yb,
                                                 const bf16_t* __restrict__ wT,
                                                 const float* __restrict__ bias,
                                                 float* __restrict__ out) {
  __shared__ __align__(16) bf16_t Bs[2][64 * 64];
  const int NTB = C_ / 64;                          // 16 N-blocks
  int mtb = blockIdx.x / NTB, ntb = blockIdx.x - mtb * NTB;
  int n0 = ntb * 64;
  int tid  = threadIdx.x;
  int lane = tid & 31;
  int wave = tid >> 5;
  int row0 = mtb * 128 + wave * 16;

  int seg0 = tid, seg1 = tid + 256;
  const bf16_t* g0 = wT + (size_t)(n0 + (seg0 >> 3)) * C_ + (seg0 & 7) * 8;
  const bf16_t* g1 = wT + (size_t)(n0 + (seg1 >> 3)) * C_ + (seg1 & 7) * 8;
  unsigned l0 = (unsigned)((seg0 >> 3) * 128 + (seg0 & 7) * 16);
  unsigned l1 = (unsigned)((seg1 >> 3) * 128 + (seg1 & 7) * 16);
  unsigned bsbase = (unsigned)(uintptr_t)(&Bs[0][0]);
  const unsigned bufoff[2] = {bsbase, bsbase + 64u * 64u * 2u};

  const int NCH = C_ / 64;
  async_copy_b128(bufoff[0] + l0, g0);
  async_copy_b128(bufoff[0] + l1, g1);

  f32x8 acc0 = {}, acc1 = {}, acc2 = {}, acc3 = {};
  for (int ch = 0; ch < NCH; ++ch) {
    int cur = ch & 1;
    if (ch + 1 < NCH) {
      async_copy_b128(bufoff[cur ^ 1] + l0, g0 + (ch + 1) * 64);
      async_copy_b128(bufoff[cur ^ 1] + l1, g1 + (ch + 1) * 64);
      asm volatile("s_wait_asynccnt 0x2" ::: "memory");
    } else {
      asm volatile("s_wait_asynccnt 0x0" ::: "memory");
    }
    __syncthreads();

    const bf16_t* bs = &Bs[cur][0];
    #pragma unroll
    for (int ko = 0; ko < 64; ko += 32) {
      bf16x16 a = load_frag_a(yb, C_, row0, ch * 64 + ko, lane);
      acc0 = wmma_bf16(a, load_frag_b(bs, 64,  0, ko, lane), acc0);
      acc1 = wmma_bf16(a, load_frag_b(bs, 64, 16, ko, lane), acc1);
      acc2 = wmma_bf16(a, load_frag_b(bs, 64, 32, ko, lane), acc2);
      acc3 = wmma_bf16(a, load_frag_b(bs, 64, 48, ko, lane), acc3);
    }
    __syncthreads();
  }

  int nlo = lane & 15, lh = lane >> 4;
  f32x8 accs[4] = {acc0, acc1, acc2, acc3};
  #pragma unroll
  for (int j = 0; j < 4; ++j) {
    int col = n0 + j * 16 + nlo;
    float bv = bias[col];
    #pragma unroll
    for (int r = 0; r < 8; ++r) {
      int row = row0 + r + lh * 8;
      out[(size_t)row * C_ + col] = accs[j][r] + bv;
    }
  }
}

// ---------------- host side ----------------

extern "C" void kernel_launch(void* const* d_in, const int* in_sizes, int n_in,
                              void* d_out, int out_size, void* d_ws, size_t ws_size,
                              hipStream_t stream) {
  const float* x      = (const float*)d_in[0];
  const float* W_qkv  = (const float*)d_in[1];
  const float* b_qkv  = (const float*)d_in[2];
  const float* W_proj = (const float*)d_in[3];
  const float* b_proj = (const float*)d_in[4];
  float* out = (float*)d_out;

  char* ws = (char*)d_ws;
  bf16_t* xb     = (bf16_t*)ws;  ws += (size_t)B_ * T_ * C_ * 2;        // 16 MB
  bf16_t* wqkvT  = (bf16_t*)ws;  ws += (size_t)C_ * C3_ * 2;            //  6 MB
  bf16_t* wprojT = (bf16_t*)ws;  ws += (size_t)C_ * C_ * 2;             //  2 MB
  bf16_t* qd     = (bf16_t*)ws;  ws += (size_t)B_ * H_ * T_ * D_ * 2;   // 16 MB
  bf16_t* kd     = (bf16_t*)ws;  ws += (size_t)B_ * H_ * T_ * D_ * 2;   // 16 MB
  bf16_t* vTd    = (bf16_t*)ws;  ws += (size_t)B_ * H_ * D_ * T_ * 2;   // 16 MB
  bf16_t* yb     = (bf16_t*)ws;  ws += (size_t)B_ * T_ * C_ * 2;        // 16 MB

  cast_f32_to_bf16<<<4096, 256, 0, stream>>>(x, xb, B_ * T_ * C_);
  transpose_cast<<<4096, 256, 0, stream>>>(W_qkv, wqkvT, C_, C3_);
  transpose_cast<<<4096, 256, 0, stream>>>(W_proj, wprojT, C_, C_);

  // (8192/128) M-blocks * 48 N-blocks = 3072 blocks, 8 waves each
  gemm_qkv<<<3072, 256, 0, stream>>>(xb, wqkvT, b_qkv, qd, kd, vTd);
  // 64 heads * 128 q-tiles = 8192 waves, 4 waves/block
  attn_flash<<<2048, 128, 0, stream>>>(qd, kd, vTd, yb);
  // (8192/128) M-blocks * 16 N-blocks = 1024 blocks
  gemm_proj<<<1024, 256, 0, stream>>>(yb, wprojT, b_proj, out);
}
